// CMCModel_17721035063745
// MI455X (gfx1250) — matrix operand order repaired and checked by reading, not compiled, not run on hardware
//
#include <hip/hip_runtime.h>
#include <hip/hip_bf16.h>

typedef __attribute__((ext_vector_type(16))) _Float16 v16h;
typedef __attribute__((ext_vector_type(8)))  float    v8f;

#define EPSF 1e-8f

// compile-time im2col decode: k (= ic*25 + ky*5 + kx) -> LDS row offset in the
// 8ch x 5row x 20col patch; k >= 200 -> zero-sentinel zone at [800..815].
__host__ __device__ constexpr int a_off(int k) {
    return (k < 200)
        ? ((k / 25) * 5 + ((k % 25) / 5)) * 20 + ((k % 25) % 5)
        : 800;
}

// ---------------------------------------------------------------------------
// conv 1->8, 5x5 SAME, relu.  in:(B,128,128)  w:(8,25)  out:(B,8,128,128)
// ---------------------------------------------------------------------------
__global__ void conv_1to8_relu(const float* __restrict__ in,
                               const float* __restrict__ w,
                               const float* __restrict__ bias,
                               float* __restrict__ out)
{
    int idx = blockIdx.x * blockDim.x + threadIdx.x;   // 32*8*128*128 = 2^22
    int x  = idx & 127;
    int y  = (idx >> 7) & 127;
    int oc = (idx >> 14) & 7;
    int b  = idx >> 17;
    const float* ip = in + b * 16384;
    float acc = bias[oc];
    #pragma unroll
    for (int ky = 0; ky < 5; ++ky) {
        int gy = y + ky - 2;
        if ((unsigned)gy >= 128u) continue;
        #pragma unroll
        for (int kx = 0; kx < 5; ++kx) {
            int gx = x + kx - 2;
            if ((unsigned)gx >= 128u) continue;
            acc = fmaf(ip[gy * 128 + gx], w[oc * 25 + ky * 5 + kx], acc);
        }
    }
    out[idx] = fmaxf(acc, 0.0f);
}

// ---------------------------------------------------------------------------
// conv 8->8, 5x5 SAME, relu — implicit GEMM on WMMA.
// One wave per (b, y, 16-px x-segment).  M=16 pixels, N=16 (8 oc padded),
// K = 8*25 = 200 padded to 224 (7 chunks of 32).
// Fully unrolled K loop => all im2col offsets are compile-time constants
// (two variants selected by lane half).  OOB K hits a zeroed LDS sentinel.
// Weights are staged zero-padded (16 x 224, f16) in LDS once per block.
// ---------------------------------------------------------------------------
__global__ void conv_8to8_wmma_relu(const float* __restrict__ in,
                                    const float* __restrict__ w,
                                    const float* __restrict__ bias,
                                    float* __restrict__ out)
{
    __shared__ _Float16 patch[8][816];   // per-wave patch + zero sentinel
    __shared__ _Float16 wlds[16 * 224];  // padded weights, B[k][n] as wlds[n*224+k]
    int tid  = threadIdx.x;
    int wave = tid >> 5;                 // 8 waves / block
    int lane = tid & 31;
    int tile = blockIdx.x * 8 + wave;    // 32*128*8 = 32768 tiles
    int xseg = tile & 7;
    int y    = (tile >> 3) & 127;
    int b    = tile >> 10;
    int x0   = xseg * 16;

    // stage padded weights (whole block): n<8 & k<200 real, else 0
    for (int i = tid; i < 16 * 224; i += 256) {
        int n = i / 224, k = i % 224;
        float v = (n < 8 && k < 200) ? w[n * 200 + k] : 0.0f;
        wlds[i] = (_Float16)v;
    }
    // stage input patch (f32 -> f16): [ (ic*5 + r)*20 + c ], sentinel zeroed
    const float* ip = in + b * (8 * 16384);
    for (int i = lane; i < 816; i += 32) {
        float v = 0.0f;
        if (i < 800) {
            int c  = i % 20;
            int t  = i / 20;
            int r  = t % 5;
            int ic = t / 5;
            int gy = y + r - 2;
            int gx = x0 + c - 2;
            if ((unsigned)gy < 128u && (unsigned)gx < 128u)
                v = ip[ic * 16384 + gy * 128 + gx];
        }
        patch[wave][i] = (_Float16)v;
    }
    __syncthreads();

    int m  = lane & 15;   // A-row (pixel); also column (oc) for B/D frags
    int hi = lane >> 4;
    v8f acc = {};
    #pragma unroll
    for (int ck = 0; ck < 7; ++ck) {     // K chunks of 32 — fully unrolled
        v16h a, bm;
        #pragma unroll
        for (int j = 0; j < 8; ++j) {
            // A 16-bit 16x32 layout: lanes 0-15 hold K pairs {2j | 16+2(j-4)},
            // lanes 16-31 the same +8.  Both offset variants are constants.
            const int kA   = ck * 32 + ((j < 4) ? 2 * j : 16 + 2 * (j - 4));
            const int o0l  = a_off(kA),     o0h = a_off(kA + 8);
            const int o1l  = a_off(kA + 1), o1h = a_off(kA + 9);
            int oa0 = hi ? o0h : o0l;
            int oa1 = hi ? o1h : o1l;
            a[2 * j]     = patch[wave][oa0 + m];
            a[2 * j + 1] = patch[wave][oa1 + m];
            // B 16-bit 32x16 layout: same K pairs as A {2j | 16+2(j-4)},
            // lanes 16-31 rows +8; column n = lane&15.  Adjacent pair.
            int kw = ck * 32 + ((j < 4) ? (hi * 8 + 2 * j) : (16 + hi * 8 + 2 * (j - 4)));
            bm[2 * j]     = wlds[m * 224 + kw];
            bm[2 * j + 1] = wlds[m * 224 + kw + 1];
        }
        acc = __builtin_amdgcn_wmma_f32_16x16x32_f16(
            false, a, false, bm, (short)0, acc, false, false);
    }
    // D[mm][n]: n = lane&15 (oc), mm = r + 8*hi (pixel)
    if (m < 8) {
        float bb = bias[m];
        float* op = out + ((b * 8 + m) * 128 + y) * 128 + x0;
        #pragma unroll
        for (int r = 0; r < 8; ++r)
            op[r + 8 * hi] = fmaxf(acc[r] + bb, 0.0f);
    }
}

// ---------------------------------------------------------------------------
// conv 8->1, 5x5 SAME, relu.  in:(B,8,128,128)  w:(8,25)  out:(B,128,128)
// ---------------------------------------------------------------------------
__global__ void conv_8to1_relu(const float* __restrict__ in,
                               const float* __restrict__ w,
                               const float* __restrict__ bias,
                               float* __restrict__ out)
{
    int idx = blockIdx.x * blockDim.x + threadIdx.x;   // 32*16384 = 524288
    int x = idx & 127, y = (idx >> 7) & 127, b = idx >> 14;
    const float* ip = in + b * (8 * 16384);
    float acc = bias[0];
    for (int ic = 0; ic < 8; ++ic) {
        #pragma unroll
        for (int ky = 0; ky < 5; ++ky) {
            int gy = y + ky - 2;
            if ((unsigned)gy >= 128u) continue;
            #pragma unroll
            for (int kx = 0; kx < 5; ++kx) {
                int gx = x + kx - 2;
                if ((unsigned)gx >= 128u) continue;
                acc = fmaf(ip[ic * 16384 + gy * 128 + gx], w[ic * 25 + ky * 5 + kx], acc);
            }
        }
    }
    out[idx] = fmaxf(acc, 0.0f);
}

// ---------------------------------------------------------------------------
// relu(concat(xs, adjs, axis=2)) -> maxpool 2x2 -> flat (B, 8192)
// ---------------------------------------------------------------------------
__global__ void concat_pool(const float* __restrict__ xs,
                            const float* __restrict__ adjs,
                            float* __restrict__ flat)
{
    int idx = blockIdx.x * blockDim.x + threadIdx.x;   // 32*64*128 = 262144
    int cc = idx & 127, r = (idx >> 7) & 63, b = idx >> 13;
    float mx = -1e30f;
    #pragma unroll
    for (int dr = 0; dr < 2; ++dr)
        #pragma unroll
        for (int dc = 0; dc < 2; ++dc) {
            int j = cc * 2 + dc, n = r * 2 + dr;
            float v = (j < 128) ? xs[(b * 128 + n) * 128 + j]
                                : adjs[(b * 128 + n) * 128 + (j - 128)];
            mx = fmaxf(mx, v);
        }
    flat[idx] = mx;
}

// ---------------------------------------------------------------------------
// FC1: (32 x 8192) @ (8192 x 512) + b, relu — WMMA, one wave per 16x16 tile.
// ---------------------------------------------------------------------------
__global__ void fc1_wmma_relu(const float* __restrict__ A,
                              const float* __restrict__ W,
                              const float* __restrict__ bias,
                              float* __restrict__ H)
{
    int lane = threadIdx.x;          // blockDim = 32 (one wave)
    int mt = blockIdx.x & 1;         // 2 M-tiles
    int nt = blockIdx.x >> 1;        // 32 N-tiles
    int m0 = mt * 16, n0 = nt * 16;
    int mm = lane & 15, hi = lane >> 4;
    const float* arow = A + (m0 + mm) * 8192;
    v8f acc = {};
    for (int ck = 0; ck < 256; ++ck) {
        __builtin_prefetch(&W[(ck * 32 + 96) * 512 + n0 + mm], 0, 1); // stream weights
        v16h a, bm;
        #pragma unroll
        for (int j = 0; j < 8; ++j) {
            int p = (j < 4) ? (hi * 8 + 2 * j) : (16 + hi * 8 + 2 * (j - 4));
            int k = ck * 32 + p;
            float2 av = *(const float2*)&arow[k];          // adjacent K pair
            auto ap = __builtin_amdgcn_cvt_pkrtz(av.x, av.y);  // v_cvt_pk_f16_f32
            a[2 * j]     = (_Float16)ap.x;
            a[2 * j + 1] = (_Float16)ap.y;
            int kb = ck * 32 + p;
            auto bp = __builtin_amdgcn_cvt_pkrtz(W[kb * 512 + n0 + mm],
                                                 W[(kb + 1) * 512 + n0 + mm]);
            bm[2 * j]     = (_Float16)bp.x;
            bm[2 * j + 1] = (_Float16)bp.y;
        }
        acc = __builtin_amdgcn_wmma_f32_16x16x32_f16(
            false, a, false, bm, (short)0, acc, false, false);
    }
    int n = n0 + mm;
    float bb = bias[n];
    #pragma unroll
    for (int r = 0; r < 8; ++r)
        H[(m0 + r + 8 * hi) * 512 + n] = fmaxf(acc[r] + bb, 0.0f);
}

// ---------------------------------------------------------------------------
// FC2 (relu) + FC3 -> out (32,15)
// ---------------------------------------------------------------------------
__global__ void fc23_out(const float* __restrict__ H,
                         const float* __restrict__ W2, const float* __restrict__ b2,
                         const float* __restrict__ W3, const float* __restrict__ b3,
                         float* __restrict__ out)
{
    __shared__ float h2[32][8];
    int t = threadIdx.x;                 // blockDim = 512
    if (t < 256) {
        int oc = t & 7, b = t >> 3;
        float acc = b2[oc];
        for (int k = 0; k < 512; ++k)
            acc = fmaf(H[b * 512 + k], W2[k * 8 + oc], acc);
        h2[b][oc] = fmaxf(acc, 0.0f);
    }
    __syncthreads();
    if (t < 480) {
        int o = t % 15, b = t / 15;
        float acc = b3[o];
        #pragma unroll
        for (int k = 0; k < 8; ++k)
            acc = fmaf(h2[b][k], W3[k * 15 + o], acc);
        out[b * 15 + o] = acc;
    }
}

// ---------------------------------------------------------------------------
// loss pipeline
// ---------------------------------------------------------------------------
__global__ void row_inv_norm(const float* __restrict__ a, float* __restrict__ rn)
{
    int idx = blockIdx.x * blockDim.x + threadIdx.x;   // 32*128 rows
    const float* p = a + idx * 128;
    float s = 0.0f;
    for (int f = 0; f < 128; ++f) s = fmaf(p[f], p[f], s);
    rn[idx] = 1.0f / fmaxf(sqrtf(s), EPSF);
}

__global__ void mean_normed(const float* __restrict__ a, const float* __restrict__ rn,
                            float* __restrict__ s_out)
{
    int idx = blockIdx.x * blockDim.x + threadIdx.x;   // 32*128 (b,f)
    int f = idx & 127, b = idx >> 7;
    float s = 0.0f;
    for (int n = 0; n < 128; ++n)
        s = fmaf(a[(b * 128 + n) * 128 + f], rn[b * 128 + n], s);
    s_out[idx] = s * (1.0f / 128.0f);
}

__global__ void sim_kernel(const float* __restrict__ sx, const float* __restrict__ sa,
                           float* __restrict__ sim)
{
    int idx = blockIdx.x * blockDim.x + threadIdx.x;   // 32*32
    int j = idx & 31, i = idx >> 5;
    float s = 0.0f;
    for (int f = 0; f < 128; ++f)
        s = fmaf(sx[i * 128 + f], sa[j * 128 + f], s);
    sim[i * 32 + j] = s;
}

__global__ void loss_kernel(const float* __restrict__ sim,
                            const int* __restrict__ neg1, const int* __restrict__ neg2,
                            float* __restrict__ out_loss)
{
    __shared__ float red[32];
    int i = threadIdx.x;                 // 32 threads
    float z0 = expf(sim[i * 32 + i]);
    float n1 = 0.0f, n2 = 0.0f;
    #pragma unroll
    for (int k = 0; k < 8; ++k) {
        n1 += sim[i * 32 + neg1[i * 8 + k]];
        n2 += sim[neg2[i * 8 + k] * 32 + i];
    }
    red[i] = logf(z0 / (z0 + n1)) + logf(z0 / (z0 + n2));
    __syncthreads();
    if (i == 0) {
        float s = 0.0f;
        for (int k = 0; k < 32; ++k) s += red[k];
        out_loss[0] = -(s / 32.0f);
    }
}

// ---------------------------------------------------------------------------
extern "C" void kernel_launch(void* const* d_in, const int* in_sizes, int n_in,
                              void* d_out, int out_size, void* d_ws, size_t ws_size,
                              hipStream_t stream)
{
    (void)in_sizes; (void)n_in; (void)out_size; (void)ws_size;
    const float* x    = (const float*)d_in[0];
    const float* adj  = (const float*)d_in[1];
    const int*   neg1 = (const int*)  d_in[2];
    const int*   neg2 = (const int*)  d_in[3];
    const float* cw1  = (const float*)d_in[4];
    const float* cb1  = (const float*)d_in[5];
    const float* cw2  = (const float*)d_in[6];
    const float* cb2  = (const float*)d_in[7];
    const float* cw3  = (const float*)d_in[8];
    const float* cb3  = (const float*)d_in[9];
    const float* cw4  = (const float*)d_in[10];
    const float* cb4  = (const float*)d_in[11];
    const float* cw5  = (const float*)d_in[12];
    const float* cb5  = (const float*)d_in[13];
    const float* cw6  = (const float*)d_in[14];
    const float* cb6  = (const float*)d_in[15];
    const float* W1   = (const float*)d_in[16];
    const float* b1   = (const float*)d_in[17];
    const float* W2   = (const float*)d_in[18];
    const float* b2   = (const float*)d_in[19];
    const float* W3   = (const float*)d_in[20];
    const float* b3   = (const float*)d_in[21];
    float* out = (float*)d_out;          // [0..479] = (32,15,1), [480] = loss

    // workspace layout (floats); big conv buffers reused across the two branches
    float* ws   = (float*)d_ws;
    float* c1   = ws;                    // 4,194,304  (B,8,128,128)
    float* c2   = c1 + 4194304;          // 4,194,304
    float* xs   = c2 + 4194304;          //   524,288  (B,128,128)
    float* adjs = xs + 524288;           //   524,288
    float* flat = adjs + 524288;         //   262,144  (B,8192)
    float* h1   = flat + 262144;         //    16,384  (B,512)
    float* rn   = h1 + 16384;            //     4,096
    float* sx   = rn + 4096;             //     4,096
    float* sa   = sx + 4096;             //     4,096
    float* sim  = sa + 4096;             //     1,024

    // x branch
    conv_1to8_relu     <<<16384, 256, 0, stream>>>(x,  cw1, cb1, c1);
    conv_8to8_wmma_relu<<< 4096, 256, 0, stream>>>(c1, cw2, cb2, c2);
    conv_8to1_relu     <<< 2048, 256, 0, stream>>>(c2, cw3, cb3, xs);
    // adj branch (reuses c1/c2 — stream-serialized)
    conv_1to8_relu     <<<16384, 256, 0, stream>>>(adj, cw4, cb4, c1);
    conv_8to8_wmma_relu<<< 4096, 256, 0, stream>>>(c1,  cw5, cb5, c2);
    conv_8to1_relu     <<< 2048, 256, 0, stream>>>(c2,  cw6, cb6, adjs);
    // head
    concat_pool  <<<1024, 256, 0, stream>>>(xs, adjs, flat);
    fc1_wmma_relu<<<  64,  32, 0, stream>>>(flat, W1, b1, h1);
    fc23_out     <<<   1, 512, 0, stream>>>(h1, W2, b2, W3, b3, out);
    // loss
    row_inv_norm<<<16, 256, 0, stream>>>(xs, rn);
    mean_normed <<<16, 256, 0, stream>>>(xs, rn, sx);
    row_inv_norm<<<16, 256, 0, stream>>>(adjs, rn);
    mean_normed <<<16, 256, 0, stream>>>(adjs, rn, sa);
    sim_kernel  <<< 4, 256, 0, stream>>>(sx, sa, sim);
    loss_kernel <<< 1,  32, 0, stream>>>(sim, neg1, neg2, out + 480);
}
